// Conv2dWeightModulate_747324310184
// MI455X (gfx1250) — matrix operand order repaired and checked by reading, not compiled, not run on hardware
//
#include <hip/hip_runtime.h>

// ---------------------------------------------------------------------------
// Problem constants (from reference)
// ---------------------------------------------------------------------------
#define B_    16
#define CIN_  512
#define COUT_ 512
#define DLAT_ 512
#define HW_   4096       // 64*64
#define KTOT_ 4608       // 9*512, K ordering: k = kk9*512 + i
#define PROWS_ 66        // padded image rows (= 64 + 2)
#define PPIX_  4356      // 66*66 padded pixels

#define ROWS_ 6          // tile rows incl. halo
#define COLS_ 66         // tile cols incl. halo
#define NPIX_ (ROWS_ * COLS_)   // 396

typedef __attribute__((ext_vector_type(16))) __bf16        v16bf;
typedef __attribute__((ext_vector_type(8)))  float         v8f;
typedef __attribute__((ext_vector_type(4)))  unsigned int  u32x4;
typedef __attribute__((ext_vector_type(4)))  unsigned int  v4u_t;
typedef __attribute__((ext_vector_type(8)))  int           v8i_t;
typedef __attribute__((ext_vector_type(4)))  int           v4i_t;

__device__ __forceinline__ unsigned short f2bf(float f) {
  unsigned int u = __float_as_uint(f);
  unsigned int r = u + 0x7FFFu + ((u >> 16) & 1u);   // round-to-nearest-even
  return (unsigned short)(r >> 16);
}

// ---------------------------------------------------------------------------
// Kernel 1: style MLP (tiny, fp32 VALU)
// ---------------------------------------------------------------------------
__global__ __launch_bounds__(512)
void style_kernel(const float* __restrict__ s,
                  const float* __restrict__ w0, const float* __restrict__ b0,
                  const float* __restrict__ a0,
                  const float* __restrict__ w1, const float* __restrict__ b1,
                  const float* __restrict__ a1,
                  const float* __restrict__ sw, const float* __restrict__ sb,
                  float* __restrict__ style, float* __restrict__ stylesq) {
  __shared__ float bufA[DLAT_];
  __shared__ float bufB[DLAT_];
  const int b = blockIdx.x;
  const int d = threadIdx.x;
  const float c_lin = 0.044194173824159216f;   // 1/sqrt(512)

  bufA[d] = s[b * DLAT_ + d];
  __syncthreads();
  float acc = 0.f;
  for (int j = 0; j < DLAT_; ++j) acc += bufA[j] * w0[d * DLAT_ + j];
  float z = acc * c_lin + b0[d];
  z = (z >= 0.f) ? z : a0[d] * z;
  bufB[d] = z;
  __syncthreads();
  acc = 0.f;
  for (int j = 0; j < DLAT_; ++j) acc += bufB[j] * w1[d * DLAT_ + j];
  z = acc * c_lin + b1[d];
  z = (z >= 0.f) ? z : a1[d] * z;
  bufA[d] = z;
  __syncthreads();
  acc = 0.f;
  for (int j = 0; j < DLAT_; ++j) acc += bufA[j] * sw[d * DLAT_ + j];
  float st = acc * c_lin + sb[d];
  style[b * CIN_ + d]   = st;
  stylesq[b * CIN_ + d] = st * st;
}

// ---------------------------------------------------------------------------
// Kernel 2: pack weights to bf16 (k = kk9*512+i, scaled by c_conv) + wsq
// ---------------------------------------------------------------------------
__global__ __launch_bounds__(256)
void pack_kernel(const float* __restrict__ conv_w,
                 unsigned short* __restrict__ wb,
                 float* __restrict__ wsq) {
  const int idx = blockIdx.x * 256 + threadIdx.x;   // (o,i)
  const int o = idx >> 9;
  const int i = idx & 511;
  const float cc = 0.014731391274719739f;           // 1/sqrt(512*9)
  const float* p = conv_w + (size_t)idx * 9;
  float ss = 0.f;
#pragma unroll
  for (int kk = 0; kk < 9; ++kk) {
    float v = p[kk] * cc;
    ss += v * v;
    wb[(size_t)o * KTOT_ + kk * CIN_ + i] = f2bf(v);
  }
  wsq[idx] = ss;
}

// ---------------------------------------------------------------------------
// Kernel 3: sigma_inv[b,o] = rsqrt( sum_i style^2[b,i] * wsq[o,i] + eps )
// ---------------------------------------------------------------------------
__global__ __launch_bounds__(256)
void sigma_kernel(const float* __restrict__ stylesq,
                  const float* __restrict__ wsq,
                  float* __restrict__ sigma_inv) {
  const int idx = blockIdx.x * 256 + threadIdx.x;   // (b,o)
  const int b = idx >> 9;
  const int o = idx & 511;
  const float* sq = stylesq + b * CIN_;
  const float* wq = wsq + (size_t)o * CIN_;
  float s = 0.f;
  for (int i = 0; i < CIN_; ++i) s += sq[i] * wq[i];
  sigma_inv[idx] = rsqrtf(s + 1e-8f);
}

// ---------------------------------------------------------------------------
// Kernel 4a: modulate + edge-pad + bf16, channel-innermost layout for TDM:
//   xsp[b][(y+1)*66 + (x+1)][ci] = x[b][ci][clamp y][clamp x] * style[b][ci]
// LDS-transposed so both global read and global write are coalesced.
// grid (66 padded rows, 8 ch-blocks of 64, B), block 256.
// ---------------------------------------------------------------------------
__global__ __launch_bounds__(256)
void mod_pad_kernel(const float* __restrict__ x,
                    const float* __restrict__ style,
                    unsigned short* __restrict__ xsp) {
  __shared__ unsigned short t[64 * 65];   // [ch][col], stride 65 vs bank conflicts
  const int pr = blockIdx.x;              // padded row 0..65
  const int cb = blockIdx.y * 64;
  const int b  = blockIdx.z;
  int ys = pr - 1; ys = ys < 0 ? 0 : (ys > 63 ? 63 : ys);

  const float* xb = x + ((size_t)b * CIN_ + cb) * HW_ + (ys << 6);
  const float* sv = style + b * CIN_ + cb;
  const int col  = threadIdx.x & 63;
  const int tseg = threadIdx.x >> 6;      // 0..3
#pragma unroll
  for (int j = 0; j < 16; ++j) {
    const int ch = tseg * 16 + j;
    t[ch * 65 + col] = f2bf(xb[(size_t)ch * HW_ + col] * sv[ch]);
  }
  __syncthreads();
  unsigned short* op = xsp + ((size_t)b * PPIX_ + (size_t)pr * PROWS_) * CIN_ + cb;
  const int ch2 = threadIdx.x & 63;
  for (int pc = tseg; pc < 66; pc += 4) {
    int cc = pc - 1; cc = cc < 0 ? 0 : (cc > 63 ? 63 : cc);
    op[(size_t)pc * CIN_ + ch2] = t[ch2 * 65 + cc];
  }
}

// ---------------------------------------------------------------------------
// Kernel 4b: implicit-GEMM conv, bf16 WMMA + Tensor Data Mover staging.
// Double-buffered LDS tiles (396 pixels x 32 ch) streamed by TDM while all
// 8 waves run 72 WMMAs per channel block on the other buffer.
// ---------------------------------------------------------------------------
#if defined(__HIP_DEVICE_COMPILE__) && __has_builtin(__builtin_amdgcn_tensor_load_to_lds)
#define TDM_OK 1
#else
#define TDM_OK 0
#endif

__global__ __launch_bounds__(256)
void conv_wmma_tdm_kernel(const unsigned short* __restrict__ xsp,
                          const unsigned short* __restrict__ wb,
                          const float* __restrict__ sigma_inv,
                          float* __restrict__ out) {
  __shared__ unsigned short ldsX[2][NPIX_ * 32];   // 2 x 25344 B

  const int tid  = threadIdx.x;
  const int lane = tid & 31;
  const int wv   = tid >> 5;
  const int n0   = blockIdx.x * 256;
  const int m0   = blockIdx.y * 64;
  const int b    = blockIdx.z;
  const int y0   = n0 >> 6;

  const int h  = lane >> 4;
  const int nl = lane & 15;

  v8f zero = {0.f, 0.f, 0.f, 0.f, 0.f, 0.f, 0.f, 0.f};
  v8f acc[4][2];
#pragma unroll
  for (int m = 0; m < 4; ++m)
#pragma unroll
    for (int nf = 0; nf < 2; ++nf) acc[m][nf] = zero;

  union Frag { v16bf v; u32x4 u[2]; };

  const int p0  = n0 + wv * 32 + nl;
  const int rr0 = ((p0 >> 6) - y0) + 1;
  const int xc0 = (p0 & 63) + 1;
  const int p1  = p0 + 16;
  const int rr1 = ((p1 >> 6) - y0) + 1;
  const int xc1 = (p1 & 63) + 1;

  // Tile base in padded tensor: padded pixel y0*66, all 66 cols x 6 rows
  const unsigned short* gbase =
      xsp + ((size_t)b * PPIX_ + (size_t)y0 * PROWS_) * CIN_;

#if TDM_OK
  // group1 (constant): data_size=2B; tensor_dim0=tile_dim0=32;
  // tensor_dim1=tile_dim1=396; tensor_dim0_stride=512.
  v8i_t g1;
  g1[0] = 0x00010000;              // data_size = 1 (2 bytes)
  g1[1] = (int)(32u << 16);        // tensor_dim0[15:0]
  g1[2] = (int)(396u << 16);       // tensor_dim0[31:16]=0 | tensor_dim1[15:0]
  g1[3] = (int)(32u << 16);        // tensor_dim1[31:16]=0 | tile_dim0
  g1[4] = 396;                     // tile_dim1 | tile_dim2=0
  g1[5] = 512;                     // tensor_dim0_stride[31:0]
  g1[6] = 0;
  g1[7] = 0;
  v4i_t z4 = {0, 0, 0, 0};
#if defined(__clang_major__) && (__clang_major__ >= 23)
  v8i_t z8 = {0, 0, 0, 0, 0, 0, 0, 0};
#endif

  auto tdm_issue = [&](int buf, int cblk) {
    const unsigned short* gp = gbase + (cblk << 5);
    unsigned long long ga = (unsigned long long)(size_t)gp;
    unsigned int laddr =
        (unsigned int)(unsigned long long)(size_t)&ldsX[buf][0];
    v4u_t g0;
    g0.x = 1u;                                          // count = 1 (valid)
    g0.y = laddr;                                       // lds_addr
    g0.z = (unsigned int)(ga & 0xFFFFFFFFull);          // global_addr lo
    g0.w = (unsigned int)((ga >> 32) & 0x01FFFFFFull) | 0x80000000u; // hi|type=2
#if defined(__clang_major__) && (__clang_major__ >= 23)
    __builtin_amdgcn_tensor_load_to_lds(g0, g1, z4, z4, z8, 0);
#else
    __builtin_amdgcn_tensor_load_to_lds(g0, g1, z4, z4, 0);
#endif
  };

  if (wv == 0) tdm_issue(0, 0);
#endif  // TDM_OK

  for (int cblk = 0; cblk < 16; ++cblk) {
    const int cb  = cblk << 5;
    const int buf = cblk & 1;

    __syncthreads();   // all waves done reading the buffer being refilled
#if TDM_OK
    if (wv == 0) {
      if (cblk < 15) {
        tdm_issue(buf ^ 1, cblk + 1);                // stream next tile
        __builtin_amdgcn_s_wait_tensorcnt(1);        // current tile done
      } else {
        __builtin_amdgcn_s_wait_tensorcnt(0);
      }
    }
#else
    // fallback: cooperative 8-byte copy of the tile (no TDM builtin)
    {
      const unsigned long long* src =
          (const unsigned long long*)(gbase + cb);     // rows stride 512 elems
      unsigned long long* dst = (unsigned long long*)&ldsX[buf][0];
      for (int e = tid; e < NPIX_ * 4; e += 256) {     // 396 pixels * 4 qwords
        const int y = e >> 2, q = e & 3;
        dst[y * 4 + q] = src[y * 128 + q];             // 128 qwords per pixel row
      }
    }
#endif
    __syncthreads();   // tile ready for everyone

    // ---- 9 conv taps ----
#pragma unroll
    for (int kk9 = 0; kk9 < 9; ++kk9) {
      const int dy = kk9 / 3 - 1;
      const int dx = kk9 % 3 - 1;
      const int k0 = kk9 * CIN_ + cb;

      Frag fb0, fb1;
      {
        const u32x4* pb0 = (const u32x4*)
            &ldsX[buf][((rr0 + dy) * COLS_ + xc0 + dx) * 32 + (h << 3)];
        fb0.u[0] = pb0[0];
        fb0.u[1] = pb0[2];
        const u32x4* pb1 = (const u32x4*)
            &ldsX[buf][((rr1 + dy) * COLS_ + xc1 + dx) * 32 + (h << 3)];
        fb1.u[0] = pb1[0];
        fb1.u[1] = pb1[2];
      }

#pragma unroll
      for (int m = 0; m < 4; ++m) {
        const unsigned short* wrow =
            wb + (size_t)(m0 + m * 16 + nl) * KTOT_ + k0 + h * 8;
        Frag fa;
        fa.u[0] = *(const u32x4*)(wrow);
        fa.u[1] = *(const u32x4*)(wrow + 16);
        acc[m][0] = __builtin_amdgcn_wmma_f32_16x16x32_bf16(
            false, fa.v, false, fb0.v, (short)0, acc[m][0], false, false);
        acc[m][1] = __builtin_amdgcn_wmma_f32_16x16x32_bf16(
            false, fa.v, false, fb1.v, (short)0, acc[m][1], false, false);
      }
    }
  }

  float* ob = out + (size_t)b * COUT_ * HW_;
  const float* si = sigma_inv + b * COUT_;
#pragma unroll
  for (int m = 0; m < 4; ++m) {
#pragma unroll
    for (int r = 0; r < 8; ++r) {
      const int o = m0 + m * 16 + h * 8 + r;
      const float sc = si[o];
      ob[(size_t)o * HW_ + p0] = acc[m][0][r] * sc;
      ob[(size_t)o * HW_ + p1] = acc[m][1][r] * sc;
    }
  }
}

// ---------------------------------------------------------------------------
// Kernel 4c: small-workspace fallback conv (VALU im2col fill)
// ---------------------------------------------------------------------------
__global__ __launch_bounds__(256)
void conv_wmma_kernel(const float* __restrict__ x,
                      const float* __restrict__ style,
                      const unsigned short* __restrict__ wb,
                      const float* __restrict__ sigma_inv,
                      float* __restrict__ out) {
  __shared__ unsigned short ldsX[NPIX_ * 32];

  const int tid  = threadIdx.x;
  const int lane = tid & 31;
  const int wv   = tid >> 5;
  const int n0   = blockIdx.x * 256;
  const int m0   = blockIdx.y * 64;
  const int b    = blockIdx.z;
  const int y0   = n0 >> 6;

  const int h  = lane >> 4;
  const int nl = lane & 15;

  const float* xb = x + (size_t)b * CIN_ * HW_;
  const float* sv = style + b * CIN_;

  v8f zero = {0.f, 0.f, 0.f, 0.f, 0.f, 0.f, 0.f, 0.f};
  v8f acc[4][2];
#pragma unroll
  for (int m = 0; m < 4; ++m)
#pragma unroll
    for (int nf = 0; nf < 2; ++nf) acc[m][nf] = zero;

  union Frag { v16bf v; u32x4 u[2]; };

  const int p0  = n0 + wv * 32 + nl;
  const int rr0 = ((p0 >> 6) - y0) + 1;
  const int xc0 = (p0 & 63) + 1;
  const int p1  = p0 + 16;
  const int rr1 = ((p1 >> 6) - y0) + 1;
  const int xc1 = (p1 & 63) + 1;

  for (int cblk = 0; cblk < 16; ++cblk) {
    const int cb = cblk << 5;
    __syncthreads();
#pragma unroll 4
    for (int e = tid; e < ROWS_ * 32 * 64; e += 256) {
      const int col = e & 63;
      const int ch  = (e >> 6) & 31;
      const int row = e >> 11;
      int ys = y0 - 1 + row; ys = ys < 0 ? 0 : (ys > 63 ? 63 : ys);
      const int ci = cb + ch;
      const float v = xb[(size_t)ci * HW_ + (ys << 6) + col] * sv[ci];
      ldsX[(row * COLS_ + col + 1) * 32 + ch] = f2bf(v);
    }
    for (int e = tid; e < ROWS_ * 32 * 2; e += 256) {
      const int cw  = e & 1;
      const int ch  = (e >> 1) & 31;
      const int row = e >> 6;
      int ys = y0 - 1 + row; ys = ys < 0 ? 0 : (ys > 63 ? 63 : ys);
      const int xs = cw ? 63 : 0;
      const int ci = cb + ch;
      const float v = xb[(size_t)ci * HW_ + (ys << 6) + xs] * sv[ci];
      ldsX[(row * COLS_ + cw * 65) * 32 + ch] = f2bf(v);
    }
    __syncthreads();

#pragma unroll
    for (int kk9 = 0; kk9 < 9; ++kk9) {
      const int dy = kk9 / 3 - 1;
      const int dx = kk9 % 3 - 1;
      const int k0 = kk9 * CIN_ + cb;
      Frag fb0, fb1;
      {
        const u32x4* pb0 = (const u32x4*)
            &ldsX[((rr0 + dy) * COLS_ + xc0 + dx) * 32 + (h << 3)];
        fb0.u[0] = pb0[0];
        fb0.u[1] = pb0[2];
        const u32x4* pb1 = (const u32x4*)
            &ldsX[((rr1 + dy) * COLS_ + xc1 + dx) * 32 + (h << 3)];
        fb1.u[0] = pb1[0];
        fb1.u[1] = pb1[2];
      }
#pragma unroll
      for (int m = 0; m < 4; ++m) {
        const unsigned short* wrow =
            wb + (size_t)(m0 + m * 16 + nl) * KTOT_ + k0 + h * 8;
        Frag fa;
        fa.u[0] = *(const u32x4*)(wrow);
        fa.u[1] = *(const u32x4*)(wrow + 16);
        acc[m][0] = __builtin_amdgcn_wmma_f32_16x16x32_bf16(
            false, fa.v, false, fb0.v, (short)0, acc[m][0], false, false);
        acc[m][1] = __builtin_amdgcn_wmma_f32_16x16x32_bf16(
            false, fa.v, false, fb1.v, (short)0, acc[m][1], false, false);
      }
    }
  }

  float* ob = out + (size_t)b * COUT_ * HW_;
  const float* si = sigma_inv + b * COUT_;
#pragma unroll
  for (int m = 0; m < 4; ++m) {
#pragma unroll
    for (int r = 0; r < 8; ++r) {
      const int o = m0 + m * 16 + h * 8 + r;
      const float sc = si[o];
      ob[(size_t)o * HW_ + p0] = acc[m][0][r] * sc;
      ob[(size_t)o * HW_ + p1] = acc[m][1][r] * sc;
    }
  }
}

// ---------------------------------------------------------------------------
// Launcher
// ---------------------------------------------------------------------------
extern "C" void kernel_launch(void* const* d_in, const int* in_sizes, int n_in,
                              void* d_out, int out_size, void* d_ws, size_t ws_size,
                              hipStream_t stream) {
  const float* x      = (const float*)d_in[0];
  const float* s      = (const float*)d_in[1];
  const float* map_w0 = (const float*)d_in[2];
  const float* map_b0 = (const float*)d_in[3];
  const float* pa0    = (const float*)d_in[4];
  const float* map_w1 = (const float*)d_in[5];
  const float* map_b1 = (const float*)d_in[6];
  const float* pa1    = (const float*)d_in[7];
  const float* stw    = (const float*)d_in[8];
  const float* stb    = (const float*)d_in[9];
  const float* conv_w = (const float*)d_in[10];
  float* out = (float*)d_out;

  // workspace carve-up
  const size_t WB_END    = 5865472ull;                     // bytes through wb
  const size_t XSP_BYTES = (size_t)B_ * PPIX_ * CIN_ * 2;  // 71,368,704

  float* style     = (float*)d_ws;
  float* stylesq   = style   + B_ * CIN_;
  float* sigma_inv = stylesq + B_ * CIN_;
  float* wsq       = sigma_inv + B_ * COUT_;
  unsigned short* wb  = (unsigned short*)(wsq + (size_t)COUT_ * CIN_);
  unsigned short* xsp = (unsigned short*)((char*)d_ws + WB_END);

  style_kernel<<<B_, 512, 0, stream>>>(s, map_w0, map_b0, pa0,
                                       map_w1, map_b1, pa1,
                                       stw, stb, style, stylesq);
  pack_kernel<<<(COUT_ * CIN_) / 256, 256, 0, stream>>>(conv_w, wb, wsq);
  sigma_kernel<<<(B_ * COUT_) / 256, 256, 0, stream>>>(stylesq, wsq, sigma_inv);

  if (ws_size >= WB_END + XSP_BYTES) {
    // TDM-streamed path: pre-pad + modulate to bf16, then async-tiled conv
    mod_pad_kernel<<<dim3(PROWS_, CIN_ / 64, B_), 256, 0, stream>>>(x, style, xsp);
    conv_wmma_tdm_kernel<<<dim3(HW_ / 256, COUT_ / 64, B_), 256, 0, stream>>>(
        xsp, wb, sigma_inv, out);
  } else {
    conv_wmma_kernel<<<dim3(HW_ / 256, COUT_ / 64, B_), 256, 0, stream>>>(
        x, style, wb, sigma_inv, out);
  }
}
